// MoELayer_5609227288847
// MI455X (gfx1250) — matrix operand order repaired and checked by reading, not compiled, not run on hardware
//
#include <hip/hip_runtime.h>

// Problem constants (from reference): B=2,S=1024,D=1024,E=8,K=2,F=512,FS=1024
#define T_TOK 2048
#define DDIM  1024
#define NEXP  8
#define TOPK  2
#define FDIM  512
#define FSH   1024

typedef __attribute__((ext_vector_type(16))) __bf16 v16bf;
typedef __attribute__((ext_vector_type(8)))  __bf16 bf16x8;
typedef __attribute__((ext_vector_type(4)))  __bf16 bf16x4;
typedef __attribute__((ext_vector_type(8)))  float  v8f;

#if defined(__has_builtin)
#if __has_builtin(__builtin_amdgcn_global_load_async_to_lds_b64)
#define HAVE_ASYNC_LDS 1
#endif
#endif
#ifndef HAVE_ASYNC_LDS
#define HAVE_ASYNC_LDS 0
#endif

// Builtin prototype (from clang diagnostic): first arg is
//   '__attribute__((__vector_size__(2*sizeof(int)))) int __device__ *'
// i.e. pointer-to-int2 (gcc vector style) in the global (AS1) address space.
typedef int v2i_gcc __attribute__((vector_size(8)));
typedef __attribute__((address_space(1))) v2i_gcc* g_v2i_ptr;
typedef __attribute__((address_space(3))) v2i_gcc* l_v2i_ptr;
#define AS1P64(p) ((g_v2i_ptr)(unsigned long long)(uintptr_t)(p))
#define AS3P64(p) ((l_v2i_ptr)(unsigned int)(unsigned long long)(uintptr_t)(p))

union AFrag { v16bf v; bf16x8 h[2]; };

__device__ __forceinline__ v8f vzero8() {
  v8f z;
#pragma unroll
  for (int i = 0; i < 8; ++i) z[i] = 0.0f;
  return z;
}

__device__ __forceinline__ v8f wmma_bf16(v16bf a, v16bf b, v8f c) {
  // (neg_a, A, neg_b, B, c_mod, C, reuse_a, reuse_b)
  return __builtin_amdgcn_wmma_f32_16x16x32_bf16(false, a, false, b, (short)0, c,
                                                 false, false);
}

// ---------------------------------------------------------------- fp32->bf16 (row major)
__global__ void cvt_bf16_kernel(const float* __restrict__ in,
                                __bf16* __restrict__ out, int n4) {
  int i = blockIdx.x * blockDim.x + threadIdx.x;
  int stride = gridDim.x * blockDim.x;
  for (; i < n4; i += stride) {
    float4 f = ((const float4*)in)[i];
    bf16x4 o;
    o[0] = (__bf16)f.x; o[1] = (__bf16)f.y; o[2] = (__bf16)f.z; o[3] = (__bf16)f.w;
    ((bf16x4*)out)[i] = o;
  }
}

// ---------------------------------------------------------------- fp32[K][N] -> bf16[N][K]
__global__ void cvt_tr_kernel(const float* __restrict__ in,
                              __bf16* __restrict__ out, int K, int N) {
  __shared__ __bf16 tile[32][40];
  size_t matoff = (size_t)blockIdx.z * (size_t)K * N;
  in += matoff;
  out += matoff;
  int k0 = blockIdx.y * 32, n0 = blockIdx.x * 32;
  int r = threadIdx.x >> 3;         // 0..31
  int c4 = (threadIdx.x & 7) * 4;   // 0..28
  float4 f = *(const float4*)(in + (size_t)(k0 + r) * N + n0 + c4);
  bf16x4 b;
  b[0] = (__bf16)f.x; b[1] = (__bf16)f.y; b[2] = (__bf16)f.z; b[3] = (__bf16)f.w;
  *(bf16x4*)&tile[r][c4] = b;
  __syncthreads();
  bf16x4 o;
#pragma unroll
  for (int i = 0; i < 4; ++i) o[i] = tile[c4 + i][r];
  *(bf16x4*)(out + (size_t)(n0 + r) * K + k0 + c4) = o;
}

// ---------------------------------------------------------------- router
__global__ void router_kernel(const float* __restrict__ x,
                              const float* __restrict__ Wr,
                              const float* __restrict__ bias,
                              int* __restrict__ topi, float* __restrict__ topp) {
  int wave = threadIdx.x >> 5;
  int lane = threadIdx.x & 31;
  int t = blockIdx.x * (blockDim.x >> 5) + wave;
  if (t >= T_TOK) return;
  float acc[NEXP];
#pragma unroll
  for (int e = 0; e < NEXP; ++e) acc[e] = 0.0f;
  const float* xr = x + (size_t)t * DDIM;
  for (int i = 0; i < DDIM / 32; ++i) {
    int d = i * 32 + lane;
    float xv = xr[d];
    const float4* w4 = (const float4*)(Wr + (size_t)d * NEXP);
    float4 wa = w4[0], wb = w4[1];
    acc[0] += xv * wa.x; acc[1] += xv * wa.y;
    acc[2] += xv * wa.z; acc[3] += xv * wa.w;
    acc[4] += xv * wb.x; acc[5] += xv * wb.y;
    acc[6] += xv * wb.z; acc[7] += xv * wb.w;
  }
#pragma unroll
  for (int e = 0; e < NEXP; ++e)
    for (int s = 16; s > 0; s >>= 1) acc[e] += __shfl_xor(acc[e], s, 32);
  if (lane == 0) {
    float lg[NEXP];
#pragma unroll
    for (int e = 0; e < NEXP; ++e) lg[e] = acc[e] + bias[e];
    int i0 = 0;
#pragma unroll
    for (int e = 1; e < NEXP; ++e) if (lg[e] > lg[i0]) i0 = e;
    int i1 = (i0 == 0) ? 1 : 0;
#pragma unroll
    for (int e = 0; e < NEXP; ++e)
      if (e != i0 && lg[e] > lg[i1]) i1 = e;
    float v0 = lg[i0], v1 = lg[i1];
    float p1 = __expf(v1 - v0);
    float s = 1.0f + p1;
    topi[2 * t] = i0; topi[2 * t + 1] = i1;
    topp[2 * t] = 1.0f / s; topp[2 * t + 1] = p1 / s;
  }
}

// ---------------------------------------------------------------- scan (deterministic lists)
__global__ void scan_kernel(const int* __restrict__ topi,
                            int* __restrict__ pair_rank,
                            int* __restrict__ off_out,
                            int* __restrict__ rowmap,
                            int* __restrict__ pair_row) {
  __shared__ int sdata[256];
  __shared__ int carry;
  __shared__ int soff[NEXP + 1];
  const int tid = threadIdx.x;
  for (int e = 0; e < NEXP; ++e) {
    if (tid == 0) carry = 0;
    __syncthreads();
    for (int base = 0; base < T_TOK; base += 256) {
      int t = base + tid;
      int i0 = topi[2 * t], i1 = topi[2 * t + 1];
      int k = (i0 == e) ? 0 : ((i1 == e) ? 1 : -1);
      int flag = (k >= 0) ? 1 : 0;
      sdata[tid] = flag;
      __syncthreads();
      for (int s = 1; s < 256; s <<= 1) {
        int v = (tid >= s) ? sdata[tid - s] : 0;
        __syncthreads();
        sdata[tid] += v;
        __syncthreads();
      }
      if (k >= 0) pair_rank[2 * t + k] = carry + sdata[tid] - 1;
      __syncthreads();
      if (tid == 0) carry += sdata[255];
      __syncthreads();
    }
    if (tid == 0) soff[e + 1] = carry;
    __syncthreads();
  }
  if (tid == 0) {
    soff[0] = 0;
    for (int e = 0; e < NEXP; ++e) soff[e + 1] += soff[e];
    for (int e = 0; e <= NEXP; ++e) off_out[e] = soff[e];
  }
  __syncthreads();
  for (int t = tid; t < T_TOK; t += 256) {
#pragma unroll
    for (int k = 0; k < TOPK; ++k) {
      int e = topi[2 * t + k];
      int row = soff[e] + pair_rank[2 * t + k];
      rowmap[row] = t;
      pair_row[2 * t + k] = row;
    }
  }
}

// ---------------------------------------------------------------- GEMM shared pieces
struct ATile { __bf16 A[32][40]; };  // 80B rows, 16B aligned

__device__ __forceinline__ void fill_A(ATile& sm, const __bf16* asrc, int kt,
                                       int frow, int fcA) {
  if (asrc) {
#if HAVE_ASYNC_LDS
    __builtin_amdgcn_global_load_async_to_lds_b64(AS1P64(asrc + kt + fcA),
                                                  AS3P64(&sm.A[frow][fcA]), 0,
                                                  0);
#else
    *(bf16x4*)&sm.A[frow][fcA] = *(const bf16x4*)(asrc + kt + fcA);
#endif
  } else {
    bf16x4 z;
#pragma unroll
    for (int i = 0; i < 4; ++i) z[i] = (__bf16)0.0f;
    *(bf16x4*)&sm.A[frow][fcA] = z;
  }
#if HAVE_ASYNC_LDS
#if __has_builtin(__builtin_amdgcn_s_wait_asynccnt)
  __builtin_amdgcn_s_wait_asynccnt(0);
#else
  asm volatile("s_wait_asynccnt 0x0" ::: "memory");
#endif
#endif
}

// gate/up GEMM tile: M=32 x N=128 per block (8 waves), K-step 32.
// Wgt/Wut are TRANSPOSED: [F][K]; B fragment = two 16B row chunks at K offsets 8*half and 16+8*half.
__device__ __forceinline__ void gateup_tile(
    ATile& sm, const __bf16* __restrict__ X, const int* __restrict__ rowmap,
    int nrows, int mb, const __bf16* __restrict__ Wgt,
    const __bf16* __restrict__ Wut, int fb, __bf16* __restrict__ H, int ldH,
    int Kdim) {
  const int tid = threadIdx.x;
  const int wv = tid >> 5, lane = tid & 31, half = lane >> 4, mr = lane & 15;
  const int frow = tid >> 3;       // 0..31
  const int fcA = (tid & 7) * 4;   // A tile col
  int grA = mb + frow;
  const __bf16* asrc = nullptr;
  if (grA < nrows) {
    int tok = rowmap ? rowmap[grA] : grA;
    asrc = X + (size_t)tok * Kdim;
  }
  const __bf16* bgrow = Wgt + (size_t)(fb + wv * 16 + mr) * Kdim;
  const __bf16* burow = Wut + (size_t)(fb + wv * 16 + mr) * Kdim;
  v8f g0 = vzero8(), g1 = vzero8(), u0 = vzero8(), u1 = vzero8();
  for (int kt = 0; kt < Kdim; kt += 32) {
    __syncthreads();
    fill_A(sm, asrc, kt, frow, fcA);
    __syncthreads();
    AFrag a0, a1;
    a0.h[0] = *(const bf16x8*)&sm.A[mr][half * 8];
    a0.h[1] = *(const bf16x8*)&sm.A[mr][16 + half * 8];
    a1.h[0] = *(const bf16x8*)&sm.A[16 + mr][half * 8];
    a1.h[1] = *(const bf16x8*)&sm.A[16 + mr][16 + half * 8];
    // B fragments: element i<8 <-> K = 8*half + i, i>=8 <-> K = 16 + 8*half + (i-8)
    AFrag bgf, buf;
    bgf.h[0] = *(const bf16x8*)(bgrow + kt + half * 8);
    bgf.h[1] = *(const bf16x8*)(bgrow + kt + 16 + half * 8);
    buf.h[0] = *(const bf16x8*)(burow + kt + half * 8);
    buf.h[1] = *(const bf16x8*)(burow + kt + 16 + half * 8);
    v16bf bg = bgf.v;
    v16bf bu = buf.v;
    g0 = wmma_bf16(a0.v, bg, g0);
    g1 = wmma_bf16(a1.v, bg, g1);
    u0 = wmma_bf16(a0.v, bu, u0);
    u1 = wmma_bf16(a1.v, bu, u1);
    if (kt + 32 < Kdim) {
      __builtin_prefetch(bgrow + kt + 32, 0, 1);
      __builtin_prefetch(burow + kt + 32, 0, 1);
    }
  }
  int n = fb + wv * 16 + mr;
#pragma unroll
  for (int r = 0; r < 8; ++r) {
    int m = r + half * 8;
    int g0r = mb + m, g1r = mb + 16 + m;
    if (g0r < nrows) {
      float gv = g0[r], uv = u0[r];
      float h = (gv / (1.0f + __expf(-gv))) * uv;
      H[(size_t)g0r * ldH + n] = (__bf16)h;
    }
    if (g1r < nrows) {
      float gv = g1[r], uv = u1[r];
      float h = (gv / (1.0f + __expf(-gv))) * uv;
      H[(size_t)g1r * ldH + n] = (__bf16)h;
    }
  }
}

__global__ void gateup_shared_kernel(const __bf16* __restrict__ X,
                                     const __bf16* __restrict__ Sgt,
                                     const __bf16* __restrict__ Sut,
                                     __bf16* __restrict__ Hs) {
  __shared__ ATile sm;
  gateup_tile(sm, X, nullptr, T_TOK, blockIdx.x * 32, Sgt, Sut,
              blockIdx.y * 128, Hs, FSH, DDIM);
}

__global__ void gateup_expert_kernel(const __bf16* __restrict__ X,
                                     const __bf16* __restrict__ Wgt,
                                     const __bf16* __restrict__ Wut,
                                     __bf16* __restrict__ Hx,
                                     const int* __restrict__ rowmap,
                                     const int* __restrict__ off) {
  __shared__ ATile sm;
  int e = blockIdx.y;
  int fb = blockIdx.x * 128;
  int base = off[e];
  int nrows = off[e + 1] - base;
  const __bf16* wg = Wgt + (size_t)e * DDIM * FDIM;
  const __bf16* wu = Wut + (size_t)e * DDIM * FDIM;
  for (int mb = 0; mb < nrows; mb += 32)
    gateup_tile(sm, X, rowmap + base, nrows, mb, wg, wu, fb,
                Hx + (size_t)base * FDIM, FDIM, DDIM);
}

// down GEMM tile: Wt is TRANSPOSED [D][Kdim]; fp32 output rows of width DDIM.
__device__ __forceinline__ void down_tile(ATile& sm,
                                          const __bf16* __restrict__ Hmat,
                                          int nrows, int mb,
                                          const __bf16* __restrict__ Wt, int nb,
                                          float* __restrict__ Out, int Kdim) {
  const int tid = threadIdx.x;
  const int wv = tid >> 5, lane = tid & 31, half = lane >> 4, mr = lane & 15;
  const int frow = tid >> 3, fcA = (tid & 7) * 4;
  int grA = mb + frow;
  const __bf16* asrc = (grA < nrows) ? (Hmat + (size_t)grA * Kdim) : nullptr;
  const __bf16* brow = Wt + (size_t)(nb + wv * 16 + mr) * Kdim;
  v8f c0 = vzero8(), c1 = vzero8();
  for (int kt = 0; kt < Kdim; kt += 32) {
    __syncthreads();
    fill_A(sm, asrc, kt, frow, fcA);
    __syncthreads();
    AFrag a0, a1;
    a0.h[0] = *(const bf16x8*)&sm.A[mr][half * 8];
    a0.h[1] = *(const bf16x8*)&sm.A[mr][16 + half * 8];
    a1.h[0] = *(const bf16x8*)&sm.A[16 + mr][half * 8];
    a1.h[1] = *(const bf16x8*)&sm.A[16 + mr][16 + half * 8];
    AFrag bbf;
    bbf.h[0] = *(const bf16x8*)(brow + kt + half * 8);
    bbf.h[1] = *(const bf16x8*)(brow + kt + 16 + half * 8);
    v16bf bb = bbf.v;
    c0 = wmma_bf16(a0.v, bb, c0);
    c1 = wmma_bf16(a1.v, bb, c1);
    if (kt + 32 < Kdim) __builtin_prefetch(brow + kt + 32, 0, 1);
  }
  int n = nb + wv * 16 + mr;
#pragma unroll
  for (int r = 0; r < 8; ++r) {
    int m = r + half * 8;
    if (mb + m < nrows) Out[(size_t)(mb + m) * DDIM + n] = c0[r];
    if (mb + 16 + m < nrows) Out[(size_t)(mb + 16 + m) * DDIM + n] = c1[r];
  }
}

__global__ void down_shared_kernel(const __bf16* __restrict__ Hs,
                                   const __bf16* __restrict__ Sdt,
                                   float* __restrict__ Out) {
  __shared__ ATile sm;
  down_tile(sm, Hs, T_TOK, blockIdx.x * 32, Sdt, blockIdx.y * 128, Out, FSH);
}

__global__ void down_expert_kernel(const __bf16* __restrict__ Hx,
                                   const __bf16* __restrict__ Wdt,
                                   float* __restrict__ Y,
                                   const int* __restrict__ off) {
  __shared__ ATile sm;
  int e = blockIdx.y;
  int nb = blockIdx.x * 128;
  int base = off[e];
  int nrows = off[e + 1] - base;
  const __bf16* Hm = Hx + (size_t)base * FDIM;
  const __bf16* W = Wdt + (size_t)e * FDIM * DDIM;
  float* Yp = Y + (size_t)base * DDIM;
  for (int mb = 0; mb < nrows; mb += 32)
    down_tile(sm, Hm, nrows, mb, W, nb, Yp, FDIM);
}

// ---------------------------------------------------------------- combine
__global__ void combine_kernel(float* __restrict__ out,
                               const float* __restrict__ Y,
                               const int* __restrict__ pair_row,
                               const float* __restrict__ topp) {
  int idx = blockIdx.x * blockDim.x + threadIdx.x;  // over T*D/4
  int t = idx / (DDIM / 4);
  int c = idx % (DDIM / 4);
  if (t >= T_TOK) return;
  int r0 = pair_row[2 * t], r1 = pair_row[2 * t + 1];
  float p0 = topp[2 * t], p1 = topp[2 * t + 1];
  float4 o = ((float4*)out)[idx];
  float4 y0 = ((const float4*)(Y + (size_t)r0 * DDIM))[c];
  float4 y1 = ((const float4*)(Y + (size_t)r1 * DDIM))[c];
  o.x += p0 * y0.x + p1 * y1.x;
  o.y += p0 * y0.y + p1 * y1.y;
  o.z += p0 * y0.z + p1 * y1.z;
  o.w += p0 * y0.w + p1 * y1.w;
  ((float4*)out)[idx] = o;
}

// ---------------------------------------------------------------- launch
extern "C" void kernel_launch(void* const* d_in, const int* in_sizes, int n_in,
                              void* d_out, int out_size, void* d_ws,
                              size_t ws_size, hipStream_t stream) {
  (void)in_sizes; (void)n_in; (void)out_size; (void)ws_size;
  const float* x  = (const float*)d_in[0];
  const float* Wr = (const float*)d_in[1];
  const float* Wg = (const float*)d_in[2];
  const float* Wu = (const float*)d_in[3];
  const float* Wd = (const float*)d_in[4];
  const float* Sg = (const float*)d_in[5];
  const float* Su = (const float*)d_in[6];
  const float* Sd = (const float*)d_in[7];
  const float* eb = (const float*)d_in[8];
  float* out = (float*)d_out;

  char* w = (char*)d_ws;
  auto alloc = [&](size_t bytes) -> void* {
    void* p = (void*)w;
    w += (bytes + 255) & ~(size_t)255;
    return p;
  };
  __bf16* xb   = (__bf16*)alloc((size_t)T_TOK * DDIM * 2);
  __bf16* Wgt  = (__bf16*)alloc((size_t)NEXP * DDIM * FDIM * 2);  // [e][F][D]
  __bf16* Wut  = (__bf16*)alloc((size_t)NEXP * DDIM * FDIM * 2);
  __bf16* Wdt  = (__bf16*)alloc((size_t)NEXP * FDIM * DDIM * 2);  // [e][D][F]
  __bf16* Sgt  = (__bf16*)alloc((size_t)DDIM * FSH * 2);          // [FS][D]
  __bf16* Sut  = (__bf16*)alloc((size_t)DDIM * FSH * 2);
  __bf16* Sdt  = (__bf16*)alloc((size_t)FSH * DDIM * 2);          // [D][FS]
  __bf16* Hs   = (__bf16*)alloc((size_t)T_TOK * FSH * 2);
  __bf16* Hx   = (__bf16*)alloc((size_t)T_TOK * TOPK * FDIM * 2);
  float*  Y    = (float*)alloc((size_t)T_TOK * TOPK * DDIM * 4);
  int* topi      = (int*)alloc((size_t)T_TOK * TOPK * 4);
  float* topp    = (float*)alloc((size_t)T_TOK * TOPK * 4);
  int* pair_rank = (int*)alloc((size_t)T_TOK * TOPK * 4);
  int* pair_row  = (int*)alloc((size_t)T_TOK * TOPK * 4);
  int* rowmap    = (int*)alloc((size_t)T_TOK * TOPK * 4);
  int* offp      = (int*)alloc((size_t)(NEXP + 1) * 4);

  // x stays row-major (A side)
  {
    int n4 = (T_TOK * DDIM) / 4;
    cvt_bf16_kernel<<<(n4 + 255) / 256, 256, 0, stream>>>(x, xb, n4);
  }
  // weights -> bf16 transposed [N][K]
  cvt_tr_kernel<<<dim3(FDIM / 32, DDIM / 32, NEXP), 256, 0, stream>>>(Wg, Wgt,
                                                                      DDIM, FDIM);
  cvt_tr_kernel<<<dim3(FDIM / 32, DDIM / 32, NEXP), 256, 0, stream>>>(Wu, Wut,
                                                                      DDIM, FDIM);
  cvt_tr_kernel<<<dim3(DDIM / 32, FDIM / 32, NEXP), 256, 0, stream>>>(Wd, Wdt,
                                                                      FDIM, DDIM);
  cvt_tr_kernel<<<dim3(FSH / 32, DDIM / 32, 1), 256, 0, stream>>>(Sg, Sgt, DDIM,
                                                                  FSH);
  cvt_tr_kernel<<<dim3(FSH / 32, DDIM / 32, 1), 256, 0, stream>>>(Su, Sut, DDIM,
                                                                  FSH);
  cvt_tr_kernel<<<dim3(DDIM / 32, FSH / 32, 1), 256, 0, stream>>>(Sd, Sdt, FSH,
                                                                  DDIM);

  router_kernel<<<T_TOK / 8, 256, 0, stream>>>(x, Wr, eb, topi, topp);
  scan_kernel<<<1, 256, 0, stream>>>(topi, pair_rank, offp, rowmap, pair_row);

  gateup_shared_kernel<<<dim3(T_TOK / 32, FSH / 128), 256, 0, stream>>>(
      xb, Sgt, Sut, Hs);
  down_shared_kernel<<<dim3(T_TOK / 32, DDIM / 128), 256, 0, stream>>>(Hs, Sdt,
                                                                       out);

  gateup_expert_kernel<<<dim3(FDIM / 128, NEXP), 256, 0, stream>>>(
      xb, Wgt, Wut, Hx, rowmap, offp);
  down_expert_kernel<<<dim3(DDIM / 128, NEXP), 256, 0, stream>>>(Hx, Wdt, Y,
                                                                 offp);

  combine_kernel<<<(T_TOK * DDIM / 4) / 256, 256, 0, stream>>>(out, Y, pair_row,
                                                               topp);
}
